// FoldedAutoencoder_18794776887590
// MI455X (gfx1250) — compile-verified, matrix operand order checked by reading, not run on hardware
//
#include <hip/hip_runtime.h>
#include <hip/hip_bf16.h>
#include <float.h>
#include <math.h>

#define BB   64
#define SS   512
#define NHH  16
#define HID  512
#define NLL  16
#define G3   1536            // 3*HID
#define D_IN 33
#define NGRID 50
#define NKT  16              // K tiles of 32 (512/32)
#define NNT  96              // N tiles of 16 (1536/16)
#define NMT  (4*SS)          // M-tile space: (B/16)*S = 2048

typedef __attribute__((ext_vector_type(16))) _Float16 v16h;
typedef __attribute__((ext_vector_type(8)))  _Float16 v8h;
typedef __attribute__((ext_vector_type(8)))  float    v8f;

__device__ inline v8f wmma_f16(v16h a, v16h b, v8f c) {
  return __builtin_amdgcn_wmma_f32_16x16x32_f16(false, a, false, b,
                                                (short)0, c, false, false);
}

// ---------------------------------------------------------------------------
// Pack (1536x512) f32 weight into WMMA B-fragment layout, f16.
// packed[(ntg*16+kt)*512 + lane*16 + i] = W[n][k]
//   n = ntg*16 + (lane&15), k = kt*32 + (i/8)*16 + (lane>>4)*8 + (i&7)
// ---------------------------------------------------------------------------
__global__ void pack_w_kernel(const float* __restrict__ w0,
                              const float* __restrict__ w1,
                              const float* __restrict__ w2,
                              _Float16* p0, _Float16* p1, _Float16* p2) {
  int e = blockIdx.x * 256 + threadIdx.x;
  if (e >= G3 * HID) return;
  int i    = e & 15;
  int lane = (e >> 4) & 31;
  int kt   = (e >> 9) & 15;
  int ntg  = e >> 13;
  int n = ntg * 16 + (lane & 15);
  int k = kt * 32 + ((i >> 3) * 16 + (lane >> 4) * 8 + (i & 7));
  const float* src = (blockIdx.y == 0) ? w0 : (blockIdx.y == 1) ? w1 : w2;
  _Float16*    dst = (blockIdx.y == 0) ? p0 : (blockIdx.y == 1) ? p1 : p2;
  dst[e] = (_Float16)src[n * HID + k];
}

// ---------------------------------------------------------------------------
// Bitonic sort of 512 (key, idx) pairs in LDS, ascending, idx tiebreak.
// ---------------------------------------------------------------------------
__device__ inline void bitonic512(float* key, int* idx) {
  int t = threadIdx.x;
  for (int k = 2; k <= 512; k <<= 1) {
    for (int j = k >> 1; j > 0; j >>= 1) {
      __syncthreads();
      int x = t ^ j;
      if (x > t) {
        float ka = key[t], kb = key[x];
        int   ia = idx[t], ib = idx[x];
        bool gt = (ka > kb) || (ka == kb && ia > ib);
        bool up = (t & k) == 0;
        if (gt == up) { key[t] = kb; key[x] = ka; idx[t] = ib; idx[x] = ia; }
      }
    }
  }
  __syncthreads();
}

// ---------------------------------------------------------------------------
// Per-batch preprocessing: phase fold, outlier flag, smooth+shift, sort,
// embedding.  One block (512 threads) per batch.
// ---------------------------------------------------------------------------
__global__ __launch_bounds__(512)
void preprocess_kernel(const float* __restrict__ lc,
                       const unsigned char* __restrict__ mask,
                       const float* __restrict__ freq,
                       float* __restrict__ phS, float* __restrict__ magS,
                       float* __restrict__ valS, float* __restrict__ emb,
                       float* __restrict__ cnt) {
  const int b = blockIdx.x, t = threadIdx.x;
  __shared__ float ph[SS], mg[SS], sk[SS], ssig[SS];
  __shared__ int   si[SS], svld[SS];
  __shared__ float ssm[NGRID];
  __shared__ float sshift;
  __shared__ int   scount;

  float period = 2.0f / freq[b];
  float tm = lc[(b * 3 + 0) * SS + t];
  float m  = lc[(b * 3 + 1) * SS + t];
  float p  = fmodf(tm, period) / period;
  ph[t] = p; mg[t] = m;
  sk[t] = p; si[t] = t;
  if (t == 0) scount = 0;
  __syncthreads();

  bitonic512(sk, si);          // sk = sorted phase, si = perm

  // Local weighted stats over +-2 circular neighbors (Epanechnikov in dphi)
  float psC = sk[t];
  float wgt[4], wm[4];
  float wsum = 0.f, lsum = 0.f;
  int o = 0;
  #pragma unroll
  for (int dw = -2; dw <= 2; ++dw) {
    if (dw == 0) continue;
    int nb = (t + dw + SS) & (SS - 1);
    float d = sk[nb] - psC;
    float r = d - 0.5f;  r -= floorf(r);
    float dphi = r - 0.5f;
    float w = 0.75f * (1.0f - dphi * dphi);
    wgt[o] = w; wm[o] = mg[si[nb]];
    wsum += w; lsum += w * wm[o];
    ++o;
  }
  float loc = lsum / wsum;
  float vs = 0.f;
  #pragma unroll
  for (int q = 0; q < 4; ++q) { float dd = wm[q] - loc; vs += dd * dd * wgt[q]; }
  float thr = loc + 5.0f * sqrtf(vs / (wsum - 1.0f));
  ssig[si[t]] = thr;
  __syncthreads();

  int vld = (mask[b * SS + t] != 0) && !(mg[t] > ssig[t]);
  svld[t] = vld;
  __syncthreads();

  // Kernel-smoothed folded curve on a 50-point grid, argmax alignment
  if (t < NGRID) {
    float g = (float)t / (float)(NGRID - 1);
    float wk = 0.f, ws = 0.f;
    const float inv_bw2 = 1.0f / (0.1f * 0.1f);
    for (int j = 0; j < SS; ++j) {
      if (svld[j]) {
        float delta = g - ph[j];
        float K = expf((cosf(6.2831853071795864f * delta) - 1.0f) * inv_bw2);
        wk += K; ws += K * mg[j];
      }
    }
    ssm[t] = ws / wk;
  }
  __syncthreads();
  if (t == 0) {
    int am = 0; float bv = ssm[0];
    for (int i = 1; i < NGRID; ++i) if (ssm[i] > bv) { bv = ssm[i]; am = i; }
    sshift = (float)am / (float)(NGRID - 1);
  }
  __syncthreads();

  ph[t] = ph[t] - sshift;
  sk[t] = vld ? ph[t] : FLT_MAX;
  si[t] = t;
  __syncthreads();

  bitonic512(sk, si);          // sort valid-first by shifted phase (stable)

  int idx = si[t];
  float pS = ph[idx], mS = mg[idx];
  phS[b * SS + t]  = pS;
  magS[b * SS + t] = mS;
  valS[b * SS + t] = (float)svld[idx];
  atomicAdd(&scount, vld);
  __syncthreads();
  if (t == 0) cnt[b] = (float)scount;

  // Fourier embedding: [cos(2*pi*phi*k) k=1..16, sin(...), mag]
  float* e = emb + (size_t)(b * SS + t) * D_IN;
  float w0 = 6.2831853071795864f * pS;
  #pragma unroll
  for (int k = 1; k <= NHH; ++k) {
    float a = w0 * (float)k;
    e[k - 1]       = cosf(a);
    e[NHH + k - 1] = sinf(a);
  }
  e[2 * NHH] = mS;
}

// ---------------------------------------------------------------------------
// Layer-0 input projection: xW0 = emb @ W_ih0^T + b_ih0  (K=33, scalar FMA)
// Output layout: xW[((bblk*S + s)*16 + (b&15))*G3 + g]
// ---------------------------------------------------------------------------
__global__ __launch_bounds__(256)
void proj0_kernel(const float* __restrict__ emb,
                  const float* __restrict__ Wih0,
                  const float* __restrict__ bih0,
                  float* __restrict__ xW) {
  int bs = blockIdx.x;            // over B*S
  int b = bs / SS, s = bs % SS;
  __shared__ float e[D_IN];
  int t = threadIdx.x;
  if (t < D_IN) e[t] = emb[(size_t)bs * D_IN + t];
  __syncthreads();
  int bblk = b >> 4, bl = b & 15;
  size_t base = ((size_t)(bblk * SS + s) * 16 + bl) * G3;
  for (int g = t; g < G3; g += 256) {
    float acc = bih0[g];
    const float* w = Wih0 + (size_t)g * D_IN;
    #pragma unroll
    for (int d = 0; d < D_IN; ++d) acc += e[d] * w[d];
    xW[base + g] = acc;
  }
}

// ---------------------------------------------------------------------------
// Layer-1 input projection GEMM with WMMA:
//   xW1 = H0 @ W_ih1^T + b_ih1    M=NMT*16 (=B*S), N=1536, K=512
// A comes pre-packed in WMMA A-fragment layout (written by the layer-0 scan).
// One 16x16 output tile per wave, 16 K-steps of v_wmma_f32_16x16x32_f16.
// ---------------------------------------------------------------------------
__global__ __launch_bounds__(256)
void gemm_xw1_kernel(const _Float16* __restrict__ A,
                     const _Float16* __restrict__ Bw,
                     const float* __restrict__ bias,
                     float* __restrict__ xW) {
  int gwave = (blockIdx.x * 256 + threadIdx.x) >> 5;
  int lane  = threadIdx.x & 31;
  int ntg = gwave % NNT;
  int mt  = gwave / NNT;
  int l16 = lane & 15, hi = lane >> 4;

  float bb = bias[ntg * 16 + l16];
  v8f c;
  #pragma unroll
  for (int j = 0; j < 8; ++j) c[j] = bb;

  const _Float16* Ab = A  + (size_t)mt  * (NKT * 512);
  const _Float16* Bb = Bw + (size_t)ntg * (NKT * 512);
  #pragma unroll
  for (int kt = 0; kt < NKT; ++kt) {
    v16h a  = *(const v16h*)(Ab + kt * 512 + lane * 16);
    v16h bf = *(const v16h*)(Bb + kt * 512 + lane * 16);
    c = wmma_f16(a, bf, c);
  }
  size_t base = (size_t)mt * 16 * G3 + ntg * 16 + l16;
  #pragma unroll
  for (int j = 0; j < 8; ++j)
    xW[base + (size_t)(j + 8 * hi) * G3] = c[j];
}

// ---------------------------------------------------------------------------
// GRU sequential scan (one layer).  Grid = 4 blocks (16 batch rows each),
// 512 threads = 16 waves.  Per step: hW = h @ W_hh^T + b_hh via WMMA
// (each wave owns column triples {nt, nt+32, nt+64} so r/z/n fragments for
// the same hidden column are register-aligned), then GRU elementwise update.
// Layer 0 writes the hidden state in packed A-fragment layout (AhalfOut).
// Layer 1 fuses the masked mean-pool accumulation (ctxOut).
// ---------------------------------------------------------------------------
__global__ __launch_bounds__(512)
void gru_scan_kernel(const float* __restrict__ xW,
                     const _Float16* __restrict__ pWhh,
                     const float* __restrict__ bhh,
                     _Float16* __restrict__ AhalfOut,   // nullable
                     const float* __restrict__ valS,    // nullable
                     float* __restrict__ ctxOut) {      // nullable
  const int bblk = blockIdx.x;
  const int t    = threadIdx.x;
  const int wave = t >> 5, lane = t & 31;
  const int l16 = lane & 15, hi = lane >> 4;

  __shared__ _Float16 hhalf[16 * HID];   // WMMA A source
  __shared__ float    hf32[16 * HID];    // f32 master copy of h
  for (int i = t; i < 16 * HID; i += 512) {
    hhalf[i] = (_Float16)0.f;
    hf32[i]  = 0.f;
  }
  __syncthreads();

  const int nt0 = 2 * wave, nt1 = 2 * wave + 1;
  float ctxa[2][8];
  #pragma unroll
  for (int tt = 0; tt < 2; ++tt)
    #pragma unroll
    for (int j = 0; j < 8; ++j) ctxa[tt][j] = 0.f;

  for (int s = 0; s < SS; ++s) {
    float hn[2][8];
    const int mt = bblk * SS + s;
    const size_t xbase0 = (size_t)mt * 16 * G3;

    #pragma unroll
    for (int tt = 0; tt < 2; ++tt) {
      const int nt = (tt == 0) ? nt0 : nt1;
      const int g  = nt * 16 + l16;

      float br = bhh[g], bz = bhh[512 + g], bn = bhh[1024 + g];
      v8f cr, cz, cn;
      #pragma unroll
      for (int j = 0; j < 8; ++j) { cr[j] = br; cz[j] = bz; cn[j] = bn; }

      const _Float16* Br = pWhh + (size_t)(nt)      * (NKT * 512);
      const _Float16* Bz = pWhh + (size_t)(nt + 32) * (NKT * 512);
      const _Float16* Bn = pWhh + (size_t)(nt + 64) * (NKT * 512);

      for (int kt = 0; kt < NKT; ++kt) {
        int abase = l16 * HID + kt * 32 + hi * 8;
        v8h lo  = *(const v8h*)(hhalf + abase);
        v8h hi8 = *(const v8h*)(hhalf + abase + 16);
        v16h a;
        #pragma unroll
        for (int i = 0; i < 8; ++i) { a[i] = lo[i]; a[8 + i] = hi8[i]; }
        int wb = kt * 512 + lane * 16;
        v16h wr = *(const v16h*)(Br + wb);
        v16h wz = *(const v16h*)(Bz + wb);
        v16h wn = *(const v16h*)(Bn + wb);
        cr = wmma_f16(a, wr, cr);
        cz = wmma_f16(a, wz, cz);
        cn = wmma_f16(a, wn, cn);
      }

      #pragma unroll
      for (int j = 0; j < 8; ++j) {
        const int M = j + 8 * hi;
        size_t xb = xbase0 + (size_t)M * G3;
        float xr = xW[xb + g];
        float xz = xW[xb + 512 + g];
        float xn = xW[xb + 1024 + g];
        float rg = 1.f / (1.f + expf(-(xr + cr[j])));
        float zg = 1.f / (1.f + expf(-(xz + cz[j])));
        float ng = tanhf(xn + rg * cn[j]);
        float hp = hf32[M * HID + g];
        float h  = (1.f - zg) * ng + zg * hp;
        hn[tt][j] = h;
        if (ctxOut) {
          float v = valS[(size_t)(bblk * 16 + M) * SS + s];
          ctxa[tt][j] += v * h;
        }
      }
    }

    __syncthreads();                 // all reads of h_{s-1} complete
    #pragma unroll
    for (int tt = 0; tt < 2; ++tt) {
      const int nt = (tt == 0) ? nt0 : nt1;
      const int g  = nt * 16 + l16;
      #pragma unroll
      for (int j = 0; j < 8; ++j) {
        const int M = j + 8 * hi;
        hf32[M * HID + g]  = hn[tt][j];
        hhalf[M * HID + g] = (_Float16)hn[tt][j];
      }
    }
    __syncthreads();                 // h_s visible to everyone

    if (AhalfOut) {
      // thread (kt=wave, lane) emits one 32B packed A-fragment slice of h_s
      const int kt = wave;
      _Float16* dst = AhalfOut + (size_t)mt * (NKT * 512) + kt * 512 + lane * 16;
      const int kb = kt * 32 + hi * 8;
      const _Float16* s0 = hhalf + l16 * HID + kb;
      #pragma unroll
      for (int i = 0; i < 8; ++i) { dst[i] = s0[i]; dst[8 + i] = s0[16 + i]; }
    }
  }

  if (ctxOut) {
    #pragma unroll
    for (int tt = 0; tt < 2; ++tt) {
      const int nt = (tt == 0) ? nt0 : nt1;
      #pragma unroll
      for (int j = 0; j < 8; ++j)
        ctxOut[(size_t)(bblk * 16 + j + 8 * hi) * HID + nt * 16 + l16] = ctxa[tt][j];
    }
  }
}

// ---------------------------------------------------------------------------
// Head: ctx/cnt -> GELU(ctx@Wl1^T+bl1) -> @Wl2^T+bl2.  One block per batch.
// ---------------------------------------------------------------------------
__global__ __launch_bounds__(512)
void head_kernel(const float* __restrict__ ctx, const float* __restrict__ cnt,
                 const float* __restrict__ Wl1, const float* __restrict__ bl1,
                 const float* __restrict__ Wl2, const float* __restrict__ bl2,
                 float* __restrict__ out) {
  int b = blockIdx.x, t = threadIdx.x;
  __shared__ float cn_[HID], h1[HID];
  float inv = 1.f / cnt[b];
  cn_[t] = ctx[(size_t)b * HID + t] * inv;
  __syncthreads();
  float acc = bl1[t];
  const float* w = Wl1 + (size_t)t * HID;
  for (int k = 0; k < HID; ++k) acc += cn_[k] * w[k];
  h1[t] = 0.5f * acc * (1.f + erff(acc * 0.70710678118654752f));
  __syncthreads();
  if (t < NLL) {
    float a2 = bl2[t];
    const float* w2 = Wl2 + (size_t)t * HID;
    for (int k = 0; k < HID; ++k) a2 += h1[k] * w2[k];
    out[b * NLL + t] = a2;
  }
}

// ---------------------------------------------------------------------------
extern "C" void kernel_launch(void* const* d_in, const int* in_sizes, int n_in,
                              void* d_out, int out_size, void* d_ws, size_t ws_size,
                              hipStream_t stream) {
  const float*         lc   = (const float*)d_in[0];
  const unsigned char* mask = (const unsigned char*)d_in[1];
  const float*         freq = (const float*)d_in[2];
  const float* Wih0 = (const float*)d_in[3];
  const float* bih0 = (const float*)d_in[4];
  const float* Whh0 = (const float*)d_in[5];
  const float* bhh0 = (const float*)d_in[6];
  const float* Wih1 = (const float*)d_in[7];
  const float* bih1 = (const float*)d_in[8];
  const float* Whh1 = (const float*)d_in[9];
  const float* bhh1 = (const float*)d_in[10];
  const float* Wl1  = (const float*)d_in[11];
  const float* bl1  = (const float*)d_in[12];
  const float* Wl2  = (const float*)d_in[13];
  const float* bl2  = (const float*)d_in[14];

  auto au = [](size_t x) { return (x + 255) & ~(size_t)255; };
  char* w = (char*)d_ws;
  size_t o = 0;
  float*    xW    = (float*)(w + o);    o += au((size_t)NMT * 16 * G3 * sizeof(float));
  _Float16* Ahalf = (_Float16*)(w + o); o += au((size_t)NMT * NKT * 512 * sizeof(_Float16));
  _Float16* pWhh0 = (_Float16*)(w + o); o += au((size_t)G3 * HID * sizeof(_Float16));
  _Float16* pWih1 = (_Float16*)(w + o); o += au((size_t)G3 * HID * sizeof(_Float16));
  _Float16* pWhh1 = (_Float16*)(w + o); o += au((size_t)G3 * HID * sizeof(_Float16));
  float*    phS   = (float*)(w + o);    o += au((size_t)BB * SS * sizeof(float));
  float*    magS  = (float*)(w + o);    o += au((size_t)BB * SS * sizeof(float));
  float*    valS  = (float*)(w + o);    o += au((size_t)BB * SS * sizeof(float));
  float*    emb   = (float*)(w + o);    o += au((size_t)BB * SS * D_IN * sizeof(float));
  float*    ctx   = (float*)(w + o);    o += au((size_t)BB * HID * sizeof(float));
  float*    cnt   = (float*)(w + o);    o += au((size_t)BB * sizeof(float));
  (void)ws_size; (void)in_sizes; (void)n_in; (void)out_size;

  // 1. pack recurrent/input weights into WMMA B-fragment layout (f16)
  pack_w_kernel<<<dim3((G3 * HID + 255) / 256, 3), 256, 0, stream>>>(
      Whh0, Wih1, Whh1, pWhh0, pWih1, pWhh1);

  // 2. per-batch preprocessing
  preprocess_kernel<<<BB, 512, 0, stream>>>(lc, mask, freq, phS, magS, valS, emb, cnt);

  // 3. layer-0 input projection (K=33, scalar)
  proj0_kernel<<<BB * SS, 256, 0, stream>>>(emb, Wih0, bih0, xW);

  // 4. layer-0 GRU scan (WMMA recurrence), emit packed A fragments of H0
  gru_scan_kernel<<<4, 512, 0, stream>>>(xW, pWhh0, bhh0, Ahalf, nullptr, nullptr);

  // 5. layer-1 input projection GEMM (WMMA), overwrite xW
  gemm_xw1_kernel<<<(NMT * NNT) / 8, 256, 0, stream>>>(Ahalf, pWih1, bih1, xW);

  // 6. layer-1 GRU scan (WMMA recurrence) with fused masked mean-pool
  gru_scan_kernel<<<4, 512, 0, stream>>>(xW, pWhh1, bhh1, nullptr, valS, ctx);

  // 7. head
  head_kernel<<<BB, 512, 0, stream>>>(ctx, cnt, Wl1, bl1, Wl2, bl2, (float*)d_out);
}
